// GraphSAGE_80410377716236
// MI455X (gfx1250) — matrix operand order repaired and checked by reading, not compiled, not run on hardware
//
#include <hip/hip_runtime.h>
#include <hip/hip_bf16.h>

#define DEG 16
#define D   128

typedef __attribute__((ext_vector_type(16))) __bf16 bfx16;
typedef __attribute__((ext_vector_type(8)))  __bf16 bfx8;
typedef __attribute__((ext_vector_type(8)))  float  fx8;

static __device__ __forceinline__ __bf16 f2bf(float f) { return (__bf16)f; }
static __device__ __forceinline__ float sigmoidf_(float x) { return 1.0f / (1.0f + __expf(-x)); }
static __device__ __forceinline__ float fast_tanh(float x) {
#if __has_builtin(__builtin_amdgcn_tanhf)
    return __builtin_amdgcn_tanhf(x);     // v_tanh_f32 (gfx1250 TRANS op)
#else
    return tanhf(x);
#endif
}

// ---------------------------------------------------------------------------
// Per-layer LSTM aggregator. One workgroup (8 wave32) = one 16-node M-tile.
// Weight-stationary: all 32 B fragments explicitly preloaded into VGPRs before
// the time loop (immune to the asm memory clobber). Gate accumulators ARE the
// i/f/g/o vectors for feature column wave*16+(lane&15), so the cell update is
// fully register-resident. Neighbor rows are staged global->LDS with CDNA5
// async-to-LDS DMA, double-buffered so the DMA overlaps the WMMA phase.
// ---------------------------------------------------------------------------
__global__ __launch_bounds__(256) void lstm_agg_kernel(
    const __bf16* __restrict__ hb,     // [N, D]  bf16 layer-input features
    const int*    __restrict__ nbrs,   // [N, DEG]
    const __bf16* __restrict__ Wih,    // [4D, D] bf16
    const __bf16* __restrict__ Whh,    // [4D, D] bf16
    const float*  __restrict__ bcomb,  // [4D]    b_ih + b_hh
    __bf16* __restrict__ hneigh,       // [N, D]  bf16 out (final hidden)
    int N)
{
    __shared__ __bf16 Xb[2][16][D + 8];  // double-buffered gathered x_t tile
    __shared__ __bf16 Hb[16][D + 8];     // hidden state tile (A-operand view)
    __shared__ int    NbrI[16][DEG];     // neighbor indices for whole block

    const int tid   = threadIdx.x;
    const int lane  = tid & 31;
    const int wave  = tid >> 5;
    const int half  = lane >> 4;
    const int l15   = lane & 15;
    const int node0 = blockIdx.x * 16;
    const int jcol  = wave * 16 + l15;   // feature column owned by this lane

    // ---- preload all neighbor indices (one per thread) + h0 = 0
    {
        int m = tid >> 4, tt = tid & 15;       // DEG == 16
        int node = node0 + m; if (node >= N) node = N - 1;
        NbrI[m][tt] = nbrs[(size_t)node * DEG + tt];
        int j0 = tt * 8;
        #pragma unroll
        for (int q = 0; q < 8; ++q) Hb[m][j0 + q] = f2bf(0.0f);
    }

    // ---- weight-stationary: preload this wave's 32 B fragments + biases into
    //      registers (values survive the asm "memory" clobbers in the loop).
    bfx16 wih_f[4][4], whh_f[4][4];          // [k0/32][gate]
    float bias_r[4];
    #pragma unroll
    for (int p = 0; p < 4; ++p) {
        bias_r[p] = bcomb[p * D + jcol];
        #pragma unroll
        for (int kk = 0; kk < 4; ++kk) {
            const __bf16* wi = Wih + (size_t)(p * D + jcol) * D + kk * 32 + half * 8;
            ((bfx8*)&wih_f[kk][p])[0] = *(const bfx8*)wi;
            ((bfx8*)&wih_f[kk][p])[1] = *(const bfx8*)(wi + 16);
            const __bf16* wh = Whh + (size_t)(p * D + jcol) * D + kk * 32 + half * 8;
            ((bfx8*)&whh_f[kk][p])[0] = *(const bfx8*)wh;
            ((bfx8*)&whh_f[kk][p])[1] = *(const bfx8*)(wh + 16);
        }
    }

    float creg[8];
    #pragma unroll
    for (int r = 0; r < 8; ++r) creg[r] = 0.0f;
    float hval[8];

    auto issue_gather = [&](int t) {           // 16B per lane, async into LDS
        int m = tid >> 4, j0 = (tid & 15) * 8;
        int nb = NbrI[m][t];
        const __bf16* src = hb + (size_t)nb * D + j0;
        unsigned lds_addr = (unsigned)(uintptr_t)&Xb[t & 1][m][j0];
        asm volatile("global_load_async_to_lds_b128 %0, %1, off"
                     :: "v"(lds_addr), "v"(src) : "memory");
    };

    __syncthreads();                 // NbrI + Hb init visible
    issue_gather(0);                 // prologue DMA

    for (int t = 0; t < DEG; ++t) {
        asm volatile("s_wait_asynccnt 0" ::: "memory");
        __syncthreads();             // X_t visible; Hb writes from t-1 visible

        if (t + 1 < DEG) issue_gather(t + 1);   // overlap DMA with compute

        // ---- gates: acc[p] = X_t Wih_p^T + H Whh_p^T + b_p   (p = i,f,g,o)
        fx8 acc[4];
        #pragma unroll
        for (int p = 0; p < 4; ++p) {
            float bv = bias_r[p];
            acc[p] = (fx8){bv, bv, bv, bv, bv, bv, bv, bv};
        }
        #pragma unroll
        for (int kk = 0; kk < 4; ++kk) {
            bfx16 a;
            ((bfx8*)&a)[0] = *(const bfx8*)&Xb[t & 1][l15][kk * 32 + half * 8];
            ((bfx8*)&a)[1] = *(const bfx8*)&Xb[t & 1][l15][kk * 32 + 16 + half * 8];
            #pragma unroll
            for (int p = 0; p < 4; ++p)
                acc[p] = __builtin_amdgcn_wmma_f32_16x16x32_bf16(
                             false, a, false, wih_f[kk][p], (short)0, acc[p], false, false);
        }
        #pragma unroll
        for (int kk = 0; kk < 4; ++kk) {
            bfx16 a;
            ((bfx8*)&a)[0] = *(const bfx8*)&Hb[l15][kk * 32 + half * 8];
            ((bfx8*)&a)[1] = *(const bfx8*)&Hb[l15][kk * 32 + 16 + half * 8];
            #pragma unroll
            for (int p = 0; p < 4; ++p)
                acc[p] = __builtin_amdgcn_wmma_f32_16x16x32_bf16(
                             false, a, false, whh_f[kk][p], (short)0, acc[p], false, false);
        }

        // ---- register-resident LSTM cell update (i,f,g,o torch order)
        #pragma unroll
        for (int r = 0; r < 8; ++r) {
            float gi = sigmoidf_(acc[0][r]);
            float gf = sigmoidf_(acc[1][r]);
            float gg = fast_tanh(acc[2][r]);
            float go = sigmoidf_(acc[3][r]);
            creg[r] = gf * creg[r] + gi * gg;
            hval[r] = go * fast_tanh(creg[r]);
        }

        __syncthreads();             // all Xb/Hb reads of this step complete
        #pragma unroll
        for (int r = 0; r < 8; ++r)
            Hb[half * 8 + r][jcol] = f2bf(hval[r]);   // visible after next barrier
    }
    __syncthreads();

    // ---- emit final hidden state (bf16)
    {
        int m = tid >> 4, j0 = (tid & 15) * 8;
        int node = node0 + m;
        if (node < N)
            *(bfx8*)(hneigh + (size_t)node * D + j0) = *(const bfx8*)&Hb[m][j0];
    }
}

// ---------------------------------------------------------------------------
// out = [relu]( h Wself^T + h_neigh Wneigh^T + bias )
// 8 waves x one 16-col tile; A fragments straight from global (L2-resident).
// ---------------------------------------------------------------------------
__global__ __launch_bounds__(256) void combine_kernel(
    const __bf16* __restrict__ hb,      // [N, D]
    const __bf16* __restrict__ hn,      // [N, D]
    const __bf16* __restrict__ Wself,   // [D, D] bf16
    const __bf16* __restrict__ Wneigh,  // [D, D] bf16
    const float*  __restrict__ bias,    // [D]
    float*  __restrict__ out_f32,       // nullable
    __bf16* __restrict__ out_bf,        // nullable (next-layer input)
    int do_relu, int N)
{
    const int tid   = threadIdx.x;
    const int lane  = tid & 31;
    const int ct    = tid >> 5;          // 8 waves = 8 column tiles of 16
    const int half  = lane >> 4;
    const int l15   = lane & 15;
    const int node0 = blockIdx.x * 16;
    const int col   = ct * 16 + l15;

    int rowNode = node0 + l15; if (rowNode >= N) rowNode = N - 1;
    const __bf16* hrow = hb + (size_t)rowNode * D;
    const __bf16* nrow = hn + (size_t)rowNode * D;

    float bv = bias[col];
    fx8 acc = {bv, bv, bv, bv, bv, bv, bv, bv};

    #pragma unroll
    for (int k0 = 0; k0 < D; k0 += 32) {
        bfx16 a, b;
        ((bfx8*)&a)[0] = *(const bfx8*)(hrow + k0 + half * 8);
        ((bfx8*)&a)[1] = *(const bfx8*)(hrow + k0 + 16 + half * 8);
        const __bf16* w = Wself + (size_t)col * D + k0 + half * 8;
        ((bfx8*)&b)[0] = *(const bfx8*)w;
        ((bfx8*)&b)[1] = *(const bfx8*)(w + 16);
        acc = __builtin_amdgcn_wmma_f32_16x16x32_bf16(false, a, false, b,
                                                      (short)0, acc, false, false);
    }
    #pragma unroll
    for (int k0 = 0; k0 < D; k0 += 32) {
        bfx16 a, b;
        ((bfx8*)&a)[0] = *(const bfx8*)(nrow + k0 + half * 8);
        ((bfx8*)&a)[1] = *(const bfx8*)(nrow + k0 + 16 + half * 8);
        const __bf16* w = Wneigh + (size_t)col * D + k0 + half * 8;
        ((bfx8*)&b)[0] = *(const bfx8*)w;
        ((bfx8*)&b)[1] = *(const bfx8*)(w + 16);
        acc = __builtin_amdgcn_wmma_f32_16x16x32_bf16(false, a, false, b,
                                                      (short)0, acc, false, false);
    }

    #pragma unroll
    for (int r = 0; r < 8; ++r) {
        int node = node0 + half * 8 + r;
        float v = acc[r];
        if (do_relu) v = fmaxf(v, 0.0f);
        if (node < N) {
            if (out_f32) out_f32[(size_t)node * D + col] = v;
            if (out_bf)  out_bf[(size_t)node * D + col] = f2bf(v);
        }
    }
}

// ---------------------------------------------------------------------------
// helpers: f32 -> bf16 conversion, bias combine
// ---------------------------------------------------------------------------
__global__ void f32_to_bf16_kernel(const float* __restrict__ src,
                                   __bf16* __restrict__ dst, size_t n)
{
    size_t i = (size_t)blockIdx.x * blockDim.x + threadIdx.x;
    size_t stride = (size_t)gridDim.x * blockDim.x;
    for (; i < n; i += stride) dst[i] = f2bf(src[i]);
}

__global__ void bias_comb_kernel(const float* __restrict__ a,
                                 const float* __restrict__ b,
                                 float* __restrict__ o, int n)
{
    int i = blockIdx.x * blockDim.x + threadIdx.x;
    if (i < n) o[i] = a[i] + b[i];
}

// ---------------------------------------------------------------------------
extern "C" void kernel_launch(void* const* d_in, const int* in_sizes, int n_in,
                              void* d_out, int out_size, void* d_ws, size_t ws_size,
                              hipStream_t stream)
{
    const float* g_feat = (const float*)d_in[0];
    const int*   nbrs   = (const int*)d_in[1];
    const int N = in_sizes[0] / D;

    const float *Wih[3], *Whh[3], *bih[3], *bhh[3], *Wse[3], *Wne[3], *bia[3];
    for (int l = 0; l < 3; ++l) {
        int base = 2 + 7 * l;
        Wih[l] = (const float*)d_in[base + 0];
        Whh[l] = (const float*)d_in[base + 1];
        bih[l] = (const float*)d_in[base + 2];
        bhh[l] = (const float*)d_in[base + 3];
        Wse[l] = (const float*)d_in[base + 4];
        Wne[l] = (const float*)d_in[base + 5];
        bia[l] = (const float*)d_in[base + 6];
    }

    // workspace layout (all blocks 256B-aligned)
    char* ws = (char*)d_ws;
    const size_t SB = (size_t)N * D * sizeof(__bf16);
    __bf16* hbA = (__bf16*)(ws);
    __bf16* hbB = (__bf16*)(ws + SB);
    __bf16* hnb = (__bf16*)(ws + 2 * SB);
    char* wp = ws + 3 * SB;
    __bf16 *Wih_b[3], *Whh_b[3], *Wse_b[3], *Wne_b[3];
    float* bcomb[3];
    for (int l = 0; l < 3; ++l) {
        Wih_b[l] = (__bf16*)wp; wp += (size_t)4 * D * D * 2;
        Whh_b[l] = (__bf16*)wp; wp += (size_t)4 * D * D * 2;
        Wse_b[l] = (__bf16*)wp; wp += (size_t)D * D * 2;
        Wne_b[l] = (__bf16*)wp; wp += (size_t)D * D * 2;
        bcomb[l] = (float*)wp;  wp += (size_t)4 * D * 4;
    }

    // ---- one-time (per launch) conversions to bf16
    f32_to_bf16_kernel<<<512, 256, 0, stream>>>(g_feat, hbA, (size_t)N * D);
    for (int l = 0; l < 3; ++l) {
        f32_to_bf16_kernel<<<64, 256, 0, stream>>>(Wih[l], Wih_b[l], (size_t)4 * D * D);
        f32_to_bf16_kernel<<<64, 256, 0, stream>>>(Whh[l], Whh_b[l], (size_t)4 * D * D);
        f32_to_bf16_kernel<<<16, 256, 0, stream>>>(Wse[l], Wse_b[l], (size_t)D * D);
        f32_to_bf16_kernel<<<16, 256, 0, stream>>>(Wne[l], Wne_b[l], (size_t)D * D);
        bias_comb_kernel<<<2, 256, 0, stream>>>(bih[l], bhh[l], bcomb[l], 4 * D);
    }

    const dim3 grid((N + 15) / 16), block(256);

    // ---- layer 1
    lstm_agg_kernel<<<grid, block, 0, stream>>>(hbA, nbrs, Wih_b[0], Whh_b[0], bcomb[0], hnb, N);
    combine_kernel<<<grid, block, 0, stream>>>(hbA, hnb, Wse_b[0], Wne_b[0], bia[0],
                                               nullptr, hbB, 1, N);
    // ---- layer 2
    lstm_agg_kernel<<<grid, block, 0, stream>>>(hbB, nbrs, Wih_b[1], Whh_b[1], bcomb[1], hnb, N);
    combine_kernel<<<grid, block, 0, stream>>>(hbB, hnb, Wse_b[1], Wne_b[1], bia[1],
                                               nullptr, hbA, 1, N);
    // ---- layer 3 (no relu, f32 output)
    lstm_agg_kernel<<<grid, block, 0, stream>>>(hbA, nbrs, Wih_b[2], Whh_b[2], bcomb[2], hnb, N);
    combine_kernel<<<grid, block, 0, stream>>>(hbA, hnb, Wse_b[2], Wne_b[2], bia[2],
                                               (float*)d_out, nullptr, 0, N);
}
